// TemporalAttention_77524159693258
// MI455X (gfx1250) — matrix operand order repaired and checked by reading, not compiled, not run on hardware
//
#include <hip/hip_runtime.h>
#include <hip/hip_fp16.h>

typedef _Float16 v8h  __attribute__((ext_vector_type(8)));
typedef _Float16 v16h __attribute__((ext_vector_type(16)));
typedef float    v8f  __attribute__((ext_vector_type(8)));

#define BS   4
#define FD   8
#define TT   256
#define DIN  64
#define NH   8
#define DOUT 64
#define KD   (NH * DIN)   // 512

#define XSTR 264          // Xt[d][l] row stride in halves (16B-aligned rows)
#define PSTR 264          // probs tile row stride in halves
#define WSTR (KD + 8)     // Wt[n][k] row stride in halves

static __device__ __forceinline__ v16h cat8(v8h lo, v8h hi) {
    return __builtin_shufflevector(lo, hi, 0,1,2,3,4,5,6,7,8,9,10,11,12,13,14,15);
}

// -------- Kernel 1: per (b,f,h): q/k proj, scores, softmax, probs @ X -> vals (f16, ws)
__global__ __launch_bounds__(256) void attn_vals_kernel(
    const float* __restrict__ X,     // (bs,F,T,d_in)
    const float* __restrict__ attc,  // (F,Nh,1)
    const float* __restrict__ alpha, // (F,)
    const float* __restrict__ Wq,    // (F,T,d_in,Nh,2)
    const float* __restrict__ Wk,    // (F,T,d_in,Nh,2)
    const float* __restrict__ Wkey,  // (F,2,2)
    const float* __restrict__ u,     // (F,T,2)
    _Float16* __restrict__ vals)     // (bs,F,T,Nh*d_in) f16
{
    __shared__ __align__(16) _Float16 Xt[DIN * XSTR];        // X[b,f]^T as f16
    __shared__ __align__(16) _Float16 probs[8 * 16 * PSTR];  // per-wave 16x256 tile
    __shared__ float qs[TT * 2];
    __shared__ float ks[TT * 2];
    __shared__ float us[TT * 2];

    const int blk = blockIdx.x;
    const int h = blk % NH;
    const int f = (blk / NH) % FD;
    const int b = blk / (NH * FD);
    const int tid = threadIdx.x;

    // ---- phase 1: q/k projections (length-64 dot, width 2) + X transpose to f16 LDS
    {
        const int t = tid;
        const float* xrow = X  + ((size_t)(b * FD + f) * TT + t) * DIN;
        const float* wq   = Wq + ((size_t)(f * TT + t) * DIN) * (NH * 2) + h * 2;
        const float* wk   = Wk + ((size_t)(f * TT + t) * DIN) * (NH * 2) + h * 2;
        float q0 = 0.f, q1 = 0.f, k0 = 0.f, k1 = 0.f;
        #pragma unroll 8
        for (int d = 0; d < DIN; ++d) {
            float xv = xrow[d];
            q0 += xv * wq[d * 16 + 0];
            q1 += xv * wq[d * 16 + 1];
            k0 += xv * wk[d * 16 + 0];
            k1 += xv * wk[d * 16 + 1];
            Xt[d * XSTR + t] = (_Float16)xv;
        }
        qs[t * 2 + 0] = q0; qs[t * 2 + 1] = q1;
        ks[t * 2 + 0] = k0; ks[t * 2 + 1] = k1;
        us[t * 2 + 0] = u[(f * TT + t) * 2 + 0];
        us[t * 2 + 1] = u[(f * TT + t) * 2 + 1];
    }
    __syncthreads();

    const float alphaF = alpha[f];
    const float acF    = attc[f * NH + h];
    const float v0 = -alphaF;              // v = -alpha * [1, -2*ac]
    const float v1 = 2.0f * alphaF * acF;
    const float W00 = Wkey[f * 4 + 0], W01 = Wkey[f * 4 + 1];
    const float W10 = Wkey[f * 4 + 2], W11 = Wkey[f * 4 + 3];

    const int wave = tid >> 5;
    const int lane = tid & 31;
    const int hb   = lane >> 4;    // half-block: 0 for lanes 0-15, 1 for 16-31
    const int lm   = lane & 15;
    _Float16* ptile = probs + wave * 16 * PSTR;

    for (int rep = 0; rep < 2; ++rep) {
        const int m0 = (wave + rep * 8) * 16;

        // ---- scores + softmax, one row at a time; lane handles l = lane + 32*i
        for (int r = 0; r < 16; ++r) {
            const int t = m0 + r;
            const float q0 = qs[t * 2], q1 = qs[t * 2 + 1];
            const float kt0 = ks[t * 2], kt1 = ks[t * 2 + 1];
            const float qa0 = q0 + v0, qa1 = q1 + v1;
            const float c0 = qa0 * W00 + qa1 * W10;  // coeff of (l-t)^2
            const float c1 = qa0 * W01 + qa1 * W11;  // coeff of (l-t)
            float sc[8];
            float mx = -1e30f;
            #pragma unroll
            for (int i = 0; i < 8; ++i) {
                const int l = lane + 32 * i;
                const float rel = (float)(l - t);
                float s = q0 * ks[l * 2] + q1 * ks[l * 2 + 1]
                        + kt0 * us[l * 2] + kt1 * us[l * 2 + 1]
                        + (c0 * rel + c1) * rel;
                sc[i] = s;
                mx = fmaxf(mx, s);
            }
            #pragma unroll
            for (int off = 16; off >= 1; off >>= 1)
                mx = fmaxf(mx, __shfl_xor(mx, off, 32));
            float sum = 0.f;
            #pragma unroll
            for (int i = 0; i < 8; ++i) { float e = __expf(sc[i] - mx); sc[i] = e; sum += e; }
            #pragma unroll
            for (int off = 16; off >= 1; off >>= 1)
                sum += __shfl_xor(sum, off, 32);
            const float inv = 1.0f / sum;
            #pragma unroll
            for (int i = 0; i < 8; ++i)
                ptile[r * PSTR + lane + 32 * i] = (_Float16)(sc[i] * inv);
        }

        // ---- WMMA: vals[m0:m0+16, 0:64] = probs(16x256) @ X(256x64), f32 accum
        const size_t base = ((size_t)(b * FD + f) * TT) * KD + (size_t)h * DIN;
        for (int nt = 0; nt < 4; ++nt) {
            v8f acc = {0.f, 0.f, 0.f, 0.f, 0.f, 0.f, 0.f, 0.f};
            #pragma unroll
            for (int ksi = 0; ksi < 8; ++ksi) {
                const int kk = ksi * 32;
                // A frag (16x32 f16): lane holds row lm, k = g*16 + hb*8 + j
                v8h a_lo = *(const v8h*)(ptile + lm * PSTR + kk + hb * 8);
                v8h a_hi = *(const v8h*)(ptile + lm * PSTR + kk + 16 + hb * 8);
                v16h A = cat8(a_lo, a_hi);
                // B frag (32x16 f16): lane holds col n = nt*16+lm, k = g*16 + hb*8 + j
                const _Float16* bp = Xt + (nt * 16 + lm) * XSTR + kk;
                v16h B = cat8(*(const v8h*)(bp + hb * 8), *(const v8h*)(bp + 16 + hb * 8));
                acc = __builtin_amdgcn_wmma_f32_16x16x32_f16(
                          false, A, false, B, (short)0, acc, false, false);
            }
            // D layout: vgpr vv -> row m0 + vv + 8*hb, col nt*16 + lm
            #pragma unroll
            for (int vv = 0; vv < 8; ++vv) {
                const int row = m0 + vv + 8 * hb;
                const int col = nt * 16 + lm;
                vals[base + (size_t)row * KD + col] = (_Float16)acc[vv];
            }
        }
        // ptile is wave-private; DScnt ordering covers reuse in next rep (no barrier)
    }
}

// -------- Kernel 2: per (b,f): out = vals(256x512) @ W[f](512x64) + bias
__global__ __launch_bounds__(256) void out_gemm_kernel(
    const _Float16* __restrict__ vals, // (bs,F,T,512) f16
    const float* __restrict__ W,       // (F,512,64)
    const float* __restrict__ bias,    // (F,T,64)
    float* __restrict__ out)           // (bs,F,T,64)
{
    __shared__ __align__(16) _Float16 Wt[DOUT * WSTR];  // W[f]^T as f16

    const int blk = blockIdx.x;
    const int f = blk % FD;
    const int b = blk / FD;
    const int tid = threadIdx.x;

    const float* Wf = W + (size_t)f * KD * DOUT;
    for (int i = tid; i < KD * DOUT; i += 256) {
        const int kk = i / DOUT;
        const int n  = i % DOUT;
        Wt[n * WSTR + kk] = (_Float16)Wf[i];
    }
    __syncthreads();

    const int wave = tid >> 5;
    const int lane = tid & 31;
    const int hb   = lane >> 4;
    const int lm   = lane & 15;
    const _Float16* vbase = vals + ((size_t)(b * FD + f) * TT) * KD;
    const float* bb = bias + (size_t)f * TT * DOUT;
    const size_t obase = ((size_t)(b * FD + f) * TT) * DOUT;

    for (int rep = 0; rep < 2; ++rep) {
        const int m0 = (wave + rep * 8) * 16;
        for (int nt = 0; nt < 4; ++nt) {
            v8f acc = {0.f, 0.f, 0.f, 0.f, 0.f, 0.f, 0.f, 0.f};
            #pragma unroll
            for (int ksi = 0; ksi < 16; ++ksi) {
                const int kk = ksi * 32;
                const _Float16* ap = vbase + (size_t)(m0 + lm) * KD + kk;
                v16h A = cat8(*(const v8h*)(ap + hb * 8),
                              *(const v8h*)(ap + 16 + hb * 8));
                const _Float16* bp = Wt + (nt * 16 + lm) * WSTR + kk;
                v16h B = cat8(*(const v8h*)(bp + hb * 8), *(const v8h*)(bp + 16 + hb * 8));
                acc = __builtin_amdgcn_wmma_f32_16x16x32_f16(
                          false, A, false, B, (short)0, acc, false, false);
            }
            #pragma unroll
            for (int vv = 0; vv < 8; ++vv) {
                const int row = m0 + vv + 8 * hb;
                const int col = nt * 16 + lm;
                out[obase + (size_t)row * DOUT + col] =
                    acc[vv] + bb[row * DOUT + col];
            }
        }
    }
}

extern "C" void kernel_launch(void* const* d_in, const int* in_sizes, int n_in,
                              void* d_out, int out_size, void* d_ws, size_t ws_size,
                              hipStream_t stream) {
    (void)in_sizes; (void)n_in; (void)out_size; (void)ws_size;
    const float* X     = (const float*)d_in[0];
    const float* attc  = (const float*)d_in[1];
    const float* alpha = (const float*)d_in[2];
    const float* Wq    = (const float*)d_in[3];
    const float* Wk    = (const float*)d_in[4];
    const float* Wkey  = (const float*)d_in[5];
    const float* u     = (const float*)d_in[6];
    const float* W     = (const float*)d_in[7];
    const float* bias  = (const float*)d_in[8];
    float* out = (float*)d_out;
    _Float16* vals = (_Float16*)d_ws;  // bs*F*T*512 f16 = 8.4 MB

    attn_vals_kernel<<<BS * FD * NH, 256, 0, stream>>>(
        X, attc, alpha, Wq, Wk, Wkey, u, vals);
    out_gemm_kernel<<<BS * FD, 256, 0, stream>>>(vals, W, bias, out);
}
